// BlockSparse_29721173688515
// MI455X (gfx1250) — matrix operand order repaired and checked by reading, not compiled, not run on hardware
//
#include <hip/hip_runtime.h>
#include <stdint.h>

// ---------------- problem constants ----------------
#define BS     32
#define KB     128
#define NB     128
#define NNZ    2048
#define BATCH  4096
#define XCOLS  (KB * BS)   // 4096
#define OCOLS  (NB * BS)   // 4096

typedef __attribute__((ext_vector_type(16))) __bf16 v16bf;
typedef __attribute__((ext_vector_type(8)))  float  v8f;

__device__ __forceinline__ uint16_t f32_to_bf16_rne(float f) {
    uint32_t u = __builtin_bit_cast(uint32_t, f);
    uint32_t r = u + 0x7FFFu + ((u >> 16) & 1u);   // round-to-nearest-even
    return (uint16_t)(r >> 16);
}
__device__ __forceinline__ float bf16_to_f32(uint16_t h) {
    uint32_t u = ((uint32_t)h) << 16;
    return __builtin_bit_cast(float, u);
}

// =====================================================================
// Prep 1: deterministic column binning (fixed accumulation order, no atomics)
// =====================================================================
__global__ void build_col_lists(const int* __restrict__ row_idx,
                                const int* __restrict__ col_idx,
                                int* __restrict__ counts,
                                int* __restrict__ lists) {
    const int n = threadIdx.x;
    if (n >= NB) return;
    int cnt = 0;
    for (int i = 0; i < NNZ; ++i) {
        if (col_idx[i] == n) {
            lists[n * KB + cnt] = (row_idx[i] << 16) | i;  // pack (row, blk)
            ++cnt;
        }
    }
    counts[n] = cnt;
}

// =====================================================================
// Prep 2: x (fp32) -> planar bf16 hi/lo, same row-major layout
// =====================================================================
__global__ __launch_bounds__(256)
void convert_x_kernel(const float* __restrict__ x,
                      uint16_t* __restrict__ xhi,
                      uint16_t* __restrict__ xlo) {
    const size_t i4 = ((size_t)blockIdx.x * 256 + threadIdx.x) * 4;
    const float4 v4 = *(const float4*)(x + i4);
    const float f[4] = {v4.x, v4.y, v4.z, v4.w};
    ushort4 h, l;
    uint16_t hh[4], ll[4];
    #pragma unroll
    for (int q = 0; q < 4; ++q) {
        hh[q] = f32_to_bf16_rne(f[q]);
        ll[q] = f32_to_bf16_rne(f[q] - bf16_to_f32(hh[q]));
    }
    h.x = hh[0]; h.y = hh[1]; h.z = hh[2]; h.w = hh[3];
    l.x = ll[0]; l.y = ll[1]; l.z = ll[2]; l.w = ll[3];
    *(ushort4*)(xhi + i4) = h;
    *(ushort4*)(xlo + i4) = l;
}

// =====================================================================
// Prep 3: w [blk][k][n] (fp32) -> transposed bf16 hi/lo [blk][n][k]
// =====================================================================
__global__ __launch_bounds__(256)
void convert_w_kernel(const float* __restrict__ w,
                      uint16_t* __restrict__ wthi,
                      uint16_t* __restrict__ wtlo) {
    const size_t i4 = ((size_t)blockIdx.x * 256 + threadIdx.x) * 4;
    const int blk = (int)(i4 >> 10);
    const int rem = (int)(i4 & 1023);
    const int k   = rem >> 5;
    const int nn  = rem & 31;          // multiple of 4
    const float4 v4 = *(const float4*)(w + i4);
    const float f[4] = {v4.x, v4.y, v4.z, v4.w};
    uint16_t* bh = wthi + (size_t)blk * (BS * BS);
    uint16_t* bl = wtlo + (size_t)blk * (BS * BS);
    #pragma unroll
    for (int q = 0; q < 4; ++q) {
        const uint16_t hi = f32_to_bf16_rne(f[q]);
        bh[(nn + q) * BS + k] = hi;
        bl[(nn + q) * BS + k] = f32_to_bf16_rne(f[q] - bf16_to_f32(hi));
    }
}

// =====================================================================
// Main kernel (fast path): all-bf16 operands, direct global fragment loads,
// no LDS, no barriers. Each wave32 computes a private 32x32 output tile.
// bf16x3 split (hi*hi + hi*lo + lo*hi) with f32 WMMA accumulation.
// =====================================================================
#define BTILE2 256          // 8 waves x 32 rows

union Frag { uint4 u[2]; v16bf v; };

__global__ __launch_bounds__(256)
void bsmm_bf16_kernel(const uint16_t* __restrict__ xhi,
                      const uint16_t* __restrict__ xlo,
                      const uint16_t* __restrict__ wthi,
                      const uint16_t* __restrict__ wtlo,
                      const float* __restrict__ bias,
                      const int* __restrict__ counts,
                      const int* __restrict__ lists,
                      float* __restrict__ out) {
    const int n    = blockIdx.x;                   // output block column
    const int lane = threadIdx.x & 31;
    const int wv   = threadIdx.x >> 5;             // wave 0..7
    const int m0   = blockIdx.y * BTILE2 + wv * 32; // wave's first batch row

    const int l15 = lane & 15;
    const int hh  = lane >> 4;                     // lane half (0/1)

    v8f acc[2][2] = {{{}, {}}, {{}, {}}};          // [mh][nh]

    const int cnt = counts[n];
    for (int j = 0; j < cnt; ++j) {
        const int e   = lists[n * KB + j];
        const int blk = e & 0xFFFF;
        const int r   = e >> 16;

        // ---- B fragments (32x16): N = nh*16 + l15 ; K half = 16*hh, 16 contiguous k
        const uint16_t* wbh = wthi + (size_t)blk * (BS * BS) + hh * 16;
        const uint16_t* wbl = wtlo + (size_t)blk * (BS * BS) + hh * 16;
        Frag b_hi[2], b_lo[2];
        #pragma unroll
        for (int nh = 0; nh < 2; ++nh) {
            const size_t bo = (size_t)(nh * 16 + l15) * BS;
            b_hi[nh].u[0] = *(const uint4*)(wbh + bo);
            b_hi[nh].u[1] = *(const uint4*)(wbh + bo + 8);
            b_lo[nh].u[0] = *(const uint4*)(wbl + bo);
            b_lo[nh].u[1] = *(const uint4*)(wbl + bo + 8);
        }

        // ---- A fragments (16x32): row = m0 + mh*16 + l15 ; K half = 16*hh, 16 contiguous k
        const size_t xbase = (size_t)m0 * XCOLS + (size_t)r * BS + hh * 16;
        Frag a_hi[2], a_lo[2];
        #pragma unroll
        for (int mh = 0; mh < 2; ++mh) {
            const size_t ao = xbase + (size_t)(mh * 16 + l15) * XCOLS;
            a_hi[mh].u[0] = *(const uint4*)(xhi + ao);
            a_hi[mh].u[1] = *(const uint4*)(xhi + ao + 8);
            a_lo[mh].u[0] = *(const uint4*)(xlo + ao);
            a_lo[mh].u[1] = *(const uint4*)(xlo + ao + 8);
        }

        // ---- 12 WMMAs, term-major so same-acc dependents are 4 apart
        #pragma unroll
        for (int mh = 0; mh < 2; ++mh)
            #pragma unroll
            for (int nh = 0; nh < 2; ++nh)
                acc[mh][nh] = __builtin_amdgcn_wmma_f32_16x16x32_bf16(
                    false, a_hi[mh].v, false, b_hi[nh].v, (short)0, acc[mh][nh], false, false);
        #pragma unroll
        for (int mh = 0; mh < 2; ++mh)
            #pragma unroll
            for (int nh = 0; nh < 2; ++nh)
                acc[mh][nh] = __builtin_amdgcn_wmma_f32_16x16x32_bf16(
                    false, a_hi[mh].v, false, b_lo[nh].v, (short)0, acc[mh][nh], false, false);
        #pragma unroll
        for (int mh = 0; mh < 2; ++mh)
            #pragma unroll
            for (int nh = 0; nh < 2; ++nh)
                acc[mh][nh] = __builtin_amdgcn_wmma_f32_16x16x32_bf16(
                    false, a_lo[mh].v, false, b_hi[nh].v, (short)0, acc[mh][nh], false, false);
    }

    // ---- epilogue: bias + relu + store
    // C layout: VGPR v -> M = v + 8*hh (+16*mh), N = l15 (+16*nh)
    float* optr = out + (size_t)n * BS;
    #pragma unroll
    for (int mh = 0; mh < 2; ++mh) {
        const int rbase = m0 + mh * 16 + hh * 8;
        #pragma unroll
        for (int nh = 0; nh < 2; ++nh) {
            const int col = nh * 16 + l15;
            const float bv = bias[n * BS + col];
            #pragma unroll
            for (int v = 0; v < 8; ++v) {
                float val = acc[mh][nh][v] + bv;
                val = val > 0.0f ? val : 0.0f;
                optr[(size_t)(rbase + v) * OCOLS + col] = val;
            }
        }
    }
}

// =====================================================================
// Fallback (small workspace): round-1 fused kernel — convert in-kernel via LDS
// =====================================================================
#define BTILE    128
#define NTHREADS 256
#define XSTR     40
#define WSTR     40

__global__ __launch_bounds__(NTHREADS)
void bsmm_fused_kernel(const float* __restrict__ x,
                       const float* __restrict__ w,
                       const float* __restrict__ bias,
                       const int* __restrict__ counts,
                       const int* __restrict__ lists,
                       float* __restrict__ out) {
    __shared__ uint16_t xs_hi[BTILE * XSTR];
    __shared__ uint16_t xs_lo[BTILE * XSTR];
    __shared__ uint16_t wt_hi[BS * WSTR];
    __shared__ uint16_t wt_lo[BS * WSTR];

    const int n    = blockIdx.x;
    const int b0   = blockIdx.y * BTILE;
    const int t    = threadIdx.x;
    const int lane = t & 31;
    const int wv   = t >> 5;
    const int m0   = wv * 16;

    v8f acc0 = {}; v8f acc1 = {};

    const int cnt = counts[n];
    for (int j = 0; j < cnt; ++j) {
        const int e   = lists[n * KB + j];
        const int blk = e & 0xFFFF;
        const int r   = e >> 16;

        __syncthreads();
        const float* xsrc = x + (size_t)b0 * XCOLS + (size_t)r * BS;
        #pragma unroll
        for (int c = 0; c < 4; ++c) {
            const int linear = c * 1024 + t * 4;
            const int row = linear >> 5;
            const int k   = linear & 31;
            const float4 v4 = *(const float4*)(xsrc + (size_t)row * XCOLS + k);
            const float f[4] = {v4.x, v4.y, v4.z, v4.w};
            #pragma unroll
            for (int q = 0; q < 4; ++q) {
                const uint16_t hi = f32_to_bf16_rne(f[q]);
                xs_hi[row * XSTR + k + q] = hi;
                xs_lo[row * XSTR + k + q] = f32_to_bf16_rne(f[q] - bf16_to_f32(hi));
            }
        }
        {
            const float* wsrc = w + (size_t)blk * (BS * BS);
            const int linear = t * 4;
            const int k  = linear >> 5;
            const int nn = linear & 31;
            const float4 v4 = *(const float4*)(wsrc + linear);
            const float f[4] = {v4.x, v4.y, v4.z, v4.w};
            #pragma unroll
            for (int q = 0; q < 4; ++q) {
                const uint16_t hi = f32_to_bf16_rne(f[q]);
                wt_hi[(nn + q) * WSTR + k] = hi;
                wt_lo[(nn + q) * WSTR + k] = f32_to_bf16_rne(f[q] - bf16_to_f32(hi));
            }
        }
        __syncthreads();

        const int arow = m0 + (lane & 15);
        const int ak0  = (lane >> 4) << 4;
        Frag a_hi, a_lo;
        a_hi.u[0] = *(const uint4*)&xs_hi[arow * XSTR + ak0];
        a_hi.u[1] = *(const uint4*)&xs_hi[arow * XSTR + ak0 + 8];
        a_lo.u[0] = *(const uint4*)&xs_lo[arow * XSTR + ak0];
        a_lo.u[1] = *(const uint4*)&xs_lo[arow * XSTR + ak0 + 8];

        const int bk0 = (lane >> 4) << 4;
        #pragma unroll
        for (int h = 0; h < 2; ++h) {
            const int bn = h * 16 + (lane & 15);
            Frag b_hi, b_lo;
            b_hi.u[0] = *(const uint4*)&wt_hi[bn * WSTR + bk0];
            b_hi.u[1] = *(const uint4*)&wt_hi[bn * WSTR + bk0 + 8];
            b_lo.u[0] = *(const uint4*)&wt_lo[bn * WSTR + bk0];
            b_lo.u[1] = *(const uint4*)&wt_lo[bn * WSTR + bk0 + 8];

            v8f c = (h == 0) ? acc0 : acc1;
            c = __builtin_amdgcn_wmma_f32_16x16x32_bf16(false, a_hi.v, false, b_hi.v, (short)0, c, false, false);
            c = __builtin_amdgcn_wmma_f32_16x16x32_bf16(false, a_hi.v, false, b_lo.v, (short)0, c, false, false);
            c = __builtin_amdgcn_wmma_f32_16x16x32_bf16(false, a_lo.v, false, b_hi.v, (short)0, c, false, false);
            if (h == 0) acc0 = c; else acc1 = c;
        }
    }

    const float* bptr = bias + n * BS;
    float* optr = out + (size_t)b0 * OCOLS + (size_t)n * BS;
    const int mbase = m0 + ((lane >> 4) << 3);
    #pragma unroll
    for (int h = 0; h < 2; ++h) {
        const v8f c = (h == 0) ? acc0 : acc1;
        const int col = h * 16 + (lane & 15);
        const float bv = bptr[col];
        #pragma unroll
        for (int v = 0; v < 8; ++v) {
            float val = c[v] + bv;
            val = val > 0.0f ? val : 0.0f;
            optr[(size_t)(mbase + v) * OCOLS + col] = val;
        }
    }
}

// =====================================================================
// Host launcher
// =====================================================================
extern "C" void kernel_launch(void* const* d_in, const int* in_sizes, int n_in,
                              void* d_out, int out_size, void* d_ws, size_t ws_size,
                              hipStream_t stream) {
    const float* x       = (const float*)d_in[0];
    const float* w       = (const float*)d_in[1];
    const float* bias    = (const float*)d_in[2];
    const int*   row_idx = (const int*)d_in[3];
    const int*   col_idx = (const int*)d_in[4];
    float*       out     = (float*)d_out;

    uint8_t* ws = (uint8_t*)d_ws;

    // workspace layout
    const size_t off_counts = 0;                         // 128 * 4      = 512 B
    const size_t off_lists  = 512;                       // 16384 * 4    = 64 KB
    const size_t off_xhi    = 66048;                     // 32 MB
    const size_t off_xlo    = off_xhi + (size_t)BATCH * XCOLS * 2;
    const size_t off_whi    = off_xlo + (size_t)BATCH * XCOLS * 2;
    const size_t off_wlo    = off_whi + (size_t)NNZ * BS * BS * 2;
    const size_t need       = off_wlo + (size_t)NNZ * BS * BS * 2;

    int* counts = (int*)(ws + off_counts);
    int* lists  = (int*)(ws + off_lists);

    build_col_lists<<<1, NB, 0, stream>>>(row_idx, col_idx, counts, lists);

    if (ws_size >= need) {
        uint16_t* xhi  = (uint16_t*)(ws + off_xhi);
        uint16_t* xlo  = (uint16_t*)(ws + off_xlo);
        uint16_t* wthi = (uint16_t*)(ws + off_whi);
        uint16_t* wtlo = (uint16_t*)(ws + off_wlo);

        const int xblocks = (int)(((size_t)BATCH * XCOLS / 4) / 256);   // 16384
        convert_x_kernel<<<xblocks, 256, 0, stream>>>(x, xhi, xlo);
        const int wblocks = (int)(((size_t)NNZ * BS * BS / 4) / 256);   // 2048
        convert_w_kernel<<<wblocks, 256, 0, stream>>>(w, wthi, wtlo);

        dim3 grid(NB, BATCH / BTILE2);   // 128 x 16
        bsmm_bf16_kernel<<<grid, 256, 0, stream>>>(xhi, xlo, wthi, wtlo,
                                                   bias, counts, lists, out);
    } else {
        dim3 grid(NB, BATCH / BTILE);    // 128 x 32
        bsmm_fused_kernel<<<grid, NTHREADS, 0, stream>>>(x, w, bias, counts, lists, out);
    }
}
